// MHSABlock_54803782697220
// MI455X (gfx1250) — compile-verified, matrix operand order checked by reading, not run on hardware
//
#include <hip/hip_runtime.h>
#include <hip/hip_bf16.h>

// ---------------------------------------------------------------------------
// MHSA block for MI455X (gfx1250, wave32, WMMA).
//  - bf16 WMMA (16x16x32) with f32 accumulation everywhere.
//  - Fused flash attention: S^T-tile trick makes the f32 accumulator layout
//    of S^T exactly the bf16 B-fragment layout for P in O = V * P^T
//    (zero cross-lane shuffles), exp2-domain softmax with scale*log2e folded
//    into q.
//  - Attention workgroup = 4 waves sharing double-buffered k/v panels in LDS,
//    staged with gfx1250 async global->LDS copies (ASYNCcnt) when available.
//  - GEMM kernels use 16x64 tiles per wave (A-fragment reused 4x).
// Workspace usage: 20 MB (xT 4MB | q 4MB | k 4MB | v 4MB | attnT 4MB).
// ---------------------------------------------------------------------------

typedef __attribute__((ext_vector_type(16))) __bf16 bf16x16;
typedef __attribute__((ext_vector_type(8)))  __bf16 bf16x8;
typedef __attribute__((ext_vector_type(8)))  float  f32x8;
typedef int i32x4_vs __attribute__((vector_size(16)));  // matches builtin param

static constexpr int B_  = 8;
static constexpr int C_  = 256;
static constexpr int NH_ = 8;
static constexpr int DH_ = 32;
static constexpr int L_  = 1024;

#define LOG2E_F 1.4426950408889634f

#if __has_builtin(__builtin_amdgcn_global_load_async_to_lds_b128) && \
    __has_builtin(__builtin_amdgcn_s_wait_asynccnt)
#define ASYNC_LDS_OK 1
#endif

// 16 bytes global -> LDS. Async (ASYNCcnt-tracked) when the toolchain has the
// gfx1250 builtin, otherwise synchronous staging through VGPRs.
__device__ __forceinline__ void copy16_g2l(const __bf16* g, __bf16* l) {
#ifdef ASYNC_LDS_OK
  __builtin_amdgcn_global_load_async_to_lds_b128(
      (__attribute__((address_space(1))) i32x4_vs*)g,
      (__attribute__((address_space(3))) i32x4_vs*)l, 0, 0);
#else
  *(bf16x8*)l = *(const bf16x8*)g;
#endif
}

template <int N>
__device__ __forceinline__ void wait_async() {
#ifdef ASYNC_LDS_OK
  __builtin_amdgcn_s_wait_asynccnt(N);
#endif
}

// --------------------------- fragment loaders ------------------------------
// 16-bit A-matrix 16x32 layout (ISA 7.12.2): lanes 0-15 hold K=0..7 (elems
// 0..7) and K=16..23 (elems 8..15); lanes 16-31 hold K=8..15 and K=24..31.
// B operand loads identically from rows of B^T (per-lane column of B).

__device__ __forceinline__ bf16x16 load_frag_f32(const float* __restrict__ base,
                                                 int ld, int row0, int k0) {
  int sub  = threadIdx.x & 15;
  int half = (threadIdx.x >> 4) & 1;
  const float* p = base + (size_t)(row0 + sub) * ld + k0 + half * 8;
  bf16x16 r;
#pragma unroll
  for (int e = 0; e < 8; ++e) {
    r[e]     = (__bf16)p[e];
    r[e + 8] = (__bf16)p[e + 16];
  }
  return r;
}

__device__ __forceinline__ bf16x16 load_frag_bf16(const __bf16* base,
                                                  int ld, int row0, int k0) {
  int sub  = threadIdx.x & 15;
  int half = (threadIdx.x >> 4) & 1;
  const __bf16* p = base + (size_t)(row0 + sub) * ld + k0 + half * 8;
  bf16x8 lo = *(const bf16x8*)p;
  bf16x8 hi = *(const bf16x8*)(p + 16);
  bf16x16 r;
#pragma unroll
  for (int e = 0; e < 8; ++e) { r[e] = lo[e]; r[e + 8] = hi[e]; }
  return r;
}

__device__ __forceinline__ f32x8 wmma_bf16(bf16x16 a, bf16x16 b, f32x8 c) {
  return __builtin_amdgcn_wmma_f32_16x16x32_bf16(
      /*neg_a=*/false, a, /*neg_b=*/false, b,
      /*c_mod=*/(short)0, c, /*reuse_a=*/false, /*reuse_b=*/false);
}

// --------------------- kernel 0: x[b][c][l] -> xT bf16 [b][l][c] -----------
__global__ __launch_bounds__(256) void xpose_kernel(const float* __restrict__ x,
                                                    __bf16* __restrict__ xt) {
  __shared__ float tile[32][33];
  int b  = blockIdx.z;
  int c0 = blockIdx.y * 32;
  int l0 = blockIdx.x * 32;
  int tx = threadIdx.x, ty = threadIdx.y;
  const float* xp = x + (size_t)b * C_ * L_;
#pragma unroll
  for (int i = 0; i < 4; ++i)
    tile[ty + 8 * i][tx] = xp[(size_t)(c0 + ty + 8 * i) * L_ + (l0 + tx)];
  __syncthreads();
  __bf16* xo = xt + (size_t)b * L_ * C_;
#pragma unroll
  for (int i = 0; i < 4; ++i)
    xo[(size_t)(l0 + ty + 8 * i) * C_ + (c0 + tx)] = (__bf16)tile[tx][ty + 8 * i];
}

// ------------------ kernel 1: QKV GEMM (16x64 tile) + scatter --------------
// q: [b][h][l][dh] bf16, pre-scaled by dh^-0.5 * log2(e)
// k: [b][h][l][dh] bf16
// v: [b][h][dh][l] bf16  (d-major: A operand of O = V * P^T)
__global__ __launch_bounds__(32) void qkv_kernel(
    const __bf16* __restrict__ xt, const float* __restrict__ w_qkv,
    const float* __restrict__ b_qkv, __bf16* __restrict__ q,
    __bf16* __restrict__ k, __bf16* __restrict__ v) {
  int b  = blockIdx.z;
  int o0 = blockIdx.y * 16;   // 0..768
  int l0 = blockIdx.x * 64;   // 0..1024
  const __bf16* xb = xt + (size_t)b * L_ * C_;

  f32x8 acc[4] = {};
#pragma unroll
  for (int k0 = 0; k0 < C_; k0 += 32) {
    bf16x16 af = load_frag_f32(w_qkv, C_, o0, k0);   // reused 4x
#pragma unroll
    for (int n = 0; n < 4; ++n) {
      bf16x16 bf = load_frag_bf16(xb, C_, l0 + n * 16, k0);
      acc[n] = wmma_bf16(af, bf, acc[n]);
    }
  }

  int sub  = threadIdx.x & 15;
  int half = (threadIdx.x >> 4) & 1;
  int ob   = o0 + half * 8;
  int part = o0 >> 8;                 // 0=q 1=k 2=v
  int head = (o0 & 255) >> 5;
  int db   = (o0 & 31) + half * 8;
  const float qscale = 0.17677669529663687f * LOG2E_F;

#pragma unroll
  for (int n = 0; n < 4; ++n) {
    int l = l0 + n * 16 + sub;
    if (part == 0) {
      bf16x8 outv;
#pragma unroll
      for (int r = 0; r < 8; ++r)
        outv[r] = (__bf16)((acc[n][r] + b_qkv[ob + r]) * qscale);
      *(bf16x8*)(q + ((size_t)((b * NH_ + head) * L_ + l)) * DH_ + db) = outv;
    } else if (part == 1) {
      bf16x8 outv;
#pragma unroll
      for (int r = 0; r < 8; ++r) outv[r] = (__bf16)(acc[n][r] + b_qkv[ob + r]);
      *(bf16x8*)(k + ((size_t)((b * NH_ + head) * L_ + l)) * DH_ + db) = outv;
    } else {
#pragma unroll
      for (int r = 0; r < 8; ++r)
        v[((size_t)((b * NH_ + head) * DH_ + db + r)) * L_ + l] =
            (__bf16)(acc[n][r] + b_qkv[ob + r]);
    }
  }
}

// --------------------- kernel 2: fused flash attention ---------------------
__device__ __forceinline__ void online_update(const f32x8& sA, const f32x8& sB,
                                              float& m_run, float& l_run,
                                              f32x8& oA, f32x8& oB,
                                              bf16x16& pf) {
  float mx = -__builtin_inff();
#pragma unroll
  for (int r = 0; r < 8; ++r) mx = fmaxf(mx, fmaxf(sA[r], sB[r]));
  mx = fmaxf(mx, __shfl_xor(mx, 16, 32));   // lanes n / n+16 share row i
  float m_new = fmaxf(m_run, mx);
  float corr  = exp2f(m_run - m_new);       // first iter: exp2(-inf) = 0
  float ls = 0.0f;
#pragma unroll
  for (int r = 0; r < 8; ++r) {
    float p0 = exp2f(sA[r] - m_new);
    float p1 = exp2f(sB[r] - m_new);
    ls += p0 + p1;
    pf[r]     = (__bf16)p0;
    pf[r + 8] = (__bf16)p1;
  }
  ls += __shfl_xor(ls, 16, 32);
  l_run = l_run * corr + ls;
  m_run = m_new;
#pragma unroll
  for (int r = 0; r < 8; ++r) { oA[r] *= corr; oB[r] *= corr; }
}

__device__ __forceinline__ void write_tile(__bf16* __restrict__ attnT, int b,
                                           int h, int i0, const f32x8& oA,
                                           const f32x8& oB, float l_run) {
  float inv = 1.0f / l_run;
  int sub  = threadIdx.x & 15;
  int half = (threadIdx.x >> 4) & 1;
  int i = i0 + sub;
  __bf16* outp = attnT + ((size_t)b * L_ + i) * C_ + h * DH_ + half * 8;
  bf16x8 a, c;
#pragma unroll
  for (int r = 0; r < 8; ++r) {
    a[r] = (__bf16)(oA[r] * inv);
    c[r] = (__bf16)(oB[r] * inv);
  }
  *(bf16x8*)outp        = a;   // d = half*8 + r
  *(bf16x8*)(outp + 16) = c;   // d = 16 + half*8 + r
}

// 4 waves/WG; each wave owns two 16-row i-tiles (128 rows/WG). k/v 32-j
// panels are shared via LDS, double-buffered with async global->LDS copies.
__global__ __launch_bounds__(128) void attn_kernel(
    const __bf16* __restrict__ qws, const __bf16* __restrict__ kws,
    const __bf16* __restrict__ vws, __bf16* __restrict__ attnT) {
  __shared__ __bf16 kbuf[2][32 * 32];  // [j_local][d]   (ld = 32)
  __shared__ __bf16 vbuf[2][32 * 32];  // [d][j_local]   (ld = 32)

  int b  = blockIdx.z;
  int h  = blockIdx.y;
  int ig = blockIdx.x;
  int tid = threadIdx.x;
  int wid = tid >> 5;
  int i0a = ig * 128 + wid * 32;
  int i0b = i0a + 16;

  const __bf16* q = qws + (size_t)(b * NH_ + h) * L_ * DH_;
  const __bf16* k = kws + (size_t)(b * NH_ + h) * L_ * DH_;
  const __bf16* v = vws + (size_t)(b * NH_ + h) * DH_ * L_;

  // one 16B transfer per thread per panel
  int koff = tid * 8;          // k panel is 2KB contiguous at k + jb*DH_
  int vd   = tid >> 2;         // v panel: 32 rows (d), 64B per row
  int vseg = (tid & 3) * 8;

  copy16_g2l(k + koff, &kbuf[0][koff]);
  copy16_g2l(v + (size_t)vd * L_ + vseg, &vbuf[0][vd * 32 + vseg]);

  bf16x16 qf0 = load_frag_bf16(q, DH_, i0a, 0);  // loop-invariant B operands
  bf16x16 qf1 = load_frag_bf16(q, DH_, i0b, 0);

  f32x8 oacc00 = {}, oacc01 = {}, oacc10 = {}, oacc11 = {};
  f32x8 zero = {};
  float m0 = -__builtin_inff(), l0 = 0.0f;
  float m1 = -__builtin_inff(), l1 = 0.0f;

  const int T = L_ / 32;
  for (int t = 0; t < T; ++t) {
    int cur = t & 1;
    if (t + 1 < T) {
      int jb = (t + 1) * 32;
      copy16_g2l(k + (size_t)jb * DH_ + koff, &kbuf[cur ^ 1][koff]);
      copy16_g2l(v + (size_t)vd * L_ + jb + vseg, &vbuf[cur ^ 1][vd * 32 + vseg]);
      wait_async<2>();   // in-order: stage t's 2 ops are complete
    } else {
      wait_async<0>();
    }
    __syncthreads();     // panels visible to all 4 waves

    bf16x16 kf0 = load_frag_bf16(kbuf[cur], 32, 0, 0);
    bf16x16 kf1 = load_frag_bf16(kbuf[cur], 32, 16, 0);
    f32x8 s00 = wmma_bf16(kf0, qf0, zero);   // S^T tiles for i-tile a
    f32x8 s01 = wmma_bf16(kf1, qf0, zero);
    f32x8 s10 = wmma_bf16(kf0, qf1, zero);   // ... and i-tile b
    f32x8 s11 = wmma_bf16(kf1, qf1, zero);

    bf16x16 pf0, pf1;
    online_update(s00, s01, m0, l0, oacc00, oacc01, pf0);
    online_update(s10, s11, m1, l1, oacc10, oacc11, pf1);

    bf16x16 vf0 = load_frag_bf16(vbuf[cur], 32, 0, 0);
    bf16x16 vf1 = load_frag_bf16(vbuf[cur], 32, 16, 0);
    oacc00 = wmma_bf16(vf0, pf0, oacc00);    // O^T accumulate
    oacc01 = wmma_bf16(vf1, pf0, oacc01);
    oacc10 = wmma_bf16(vf0, pf1, oacc10);
    oacc11 = wmma_bf16(vf1, pf1, oacc11);
    __syncthreads();     // all waves done with buffer before refill
  }

  write_tile(attnT, b, h, i0a, oacc00, oacc01, l0);
  write_tile(attnT, b, h, i0b, oacc10, oacc11, l1);
}

// ------------- kernel 3: output projection (16x64) + residual --------------
__global__ __launch_bounds__(32) void proj_kernel(
    const __bf16* __restrict__ attnT, const float* __restrict__ w_proj,
    const float* __restrict__ b_proj, const float* __restrict__ x,
    float* __restrict__ out) {
  int b  = blockIdx.z;
  int o0 = blockIdx.y * 16;
  int l0 = blockIdx.x * 64;
  const __bf16* ab = attnT + (size_t)b * L_ * C_;

  f32x8 acc[4] = {};
#pragma unroll
  for (int k0 = 0; k0 < C_; k0 += 32) {
    bf16x16 af = load_frag_f32(w_proj, C_, o0, k0);
#pragma unroll
    for (int n = 0; n < 4; ++n) {
      bf16x16 bf = load_frag_bf16(ab, C_, l0 + n * 16, k0);
      acc[n] = wmma_bf16(af, bf, acc[n]);
    }
  }

  int sub  = threadIdx.x & 15;
  int half = (threadIdx.x >> 4) & 1;
  int ob   = o0 + half * 8;
#pragma unroll
  for (int n = 0; n < 4; ++n) {
    int l = l0 + n * 16 + sub;
#pragma unroll
    for (int r = 0; r < 8; ++r) {
      size_t idx = ((size_t)(b * C_ + ob + r)) * L_ + l;
      out[idx] = x[idx] + acc[n][r] + b_proj[ob + r];
    }
  }
}

// ------------------------------- launcher ----------------------------------
extern "C" void kernel_launch(void* const* d_in, const int* in_sizes, int n_in,
                              void* d_out, int out_size, void* d_ws, size_t ws_size,
                              hipStream_t stream) {
  (void)in_sizes; (void)n_in; (void)out_size; (void)ws_size;
  const float* x      = (const float*)d_in[0];
  const float* w_qkv  = (const float*)d_in[1];
  const float* b_qkv  = (const float*)d_in[2];
  const float* w_proj = (const float*)d_in[3];
  const float* b_proj = (const float*)d_in[4];
  float* out = (float*)d_out;

  char* ws = (char*)d_ws;
  __bf16* xt    = (__bf16*)(ws);                         // 4 MB
  __bf16* qws   = (__bf16*)(ws + (size_t)(4u << 20));    // 4 MB
  __bf16* kws   = (__bf16*)(ws + (size_t)(8u << 20));    // 4 MB
  __bf16* vws   = (__bf16*)(ws + (size_t)(12u << 20));   // 4 MB
  __bf16* attnT = (__bf16*)(ws + (size_t)(16u << 20));   // 4 MB

  xpose_kernel<<<dim3(L_ / 32, C_ / 32, B_), dim3(32, 8), 0, stream>>>(x, xt);
  qkv_kernel<<<dim3(L_ / 64, (3 * C_) / 16, B_), dim3(32), 0, stream>>>(
      xt, w_qkv, b_qkv, qws, kws, vws);
  attn_kernel<<<dim3(L_ / 128, NH_, B_), dim3(128), 0, stream>>>(qws, kws, vws,
                                                                 attnT);
  proj_kernel<<<dim3(L_ / 64, C_ / 16, B_), dim3(32), 0, stream>>>(
      attnT, w_proj, b_proj, x, out);
}
